// PatchEmbedder2ConvLayerAR_86303072845936
// MI455X (gfx1250) — matrix operand order repaired and checked by reading, not compiled, not run on hardware
//
#include <hip/hip_runtime.h>

typedef __attribute__((ext_vector_type(2)))  float  v2f;
typedef __attribute__((ext_vector_type(8)))  float  v8f;
typedef __attribute__((ext_vector_type(8)))  __bf16 v8bf;
typedef __attribute__((ext_vector_type(16))) __bf16 v16bf;
typedef __attribute__((__vector_size__(16))) int    v4i;
typedef __attribute__((address_space(1)))    v4i    gv4i;   // global
typedef __attribute__((address_space(3)))    v4i    lv4i;   // LDS

constexpr int kNodes    = 65536;
constexpr int kGraphs   = 1024;
constexpr int kNPG      = 64;       // nodes per graph
constexpr int kEdges    = 1048576;
constexpr int kInF      = 512;
constexpr int kHid      = 1024;
constexpr int kH4       = 256;
constexpr int kEmb      = 512;
constexpr int kCat      = kInF + kHid + kH4;   // 1792
constexpr float kEps    = 1e-5f;
constexpr float kSlope  = 0.01f;

// LDS B-tile row stride (elements). 40*2B = 80B: 16B-aligned rows, and the
// per-lane ds_load_b128 stride of 20 banks is conflict-free across 16 lanes.
constexpr int kBStride  = 40;

__device__ __forceinline__ float leaky_f(float x) { return x >= 0.f ? x : kSlope * x; }

__device__ __forceinline__ void split_bf16(float x, __bf16& hi, __bf16& lo) {
    hi = (__bf16)x;
    lo = (__bf16)(x - (float)hi);
}

// ---- 16B global->LDS staging: async copy if available, VGPR bounce otherwise ----
__device__ __forceinline__ void stage16(const __bf16* g, __bf16* l) {
#if __has_builtin(__builtin_amdgcn_global_load_async_to_lds_b128)
    __builtin_amdgcn_global_load_async_to_lds_b128((gv4i*)g, (lv4i*)l, 0, 0);
#else
    *(v8bf*)l = *(const v8bf*)g;
#endif
}
__device__ __forceinline__ void stage_wait() {
#if __has_builtin(__builtin_amdgcn_global_load_async_to_lds_b128)
#if __has_builtin(__builtin_amdgcn_s_wait_asynccnt)
    __builtin_amdgcn_s_wait_asynccnt(0);
#else
    asm volatile("s_wait_asynccnt 0x0" ::: "memory");
#endif
#endif
}

// =================== bf16 16x16x32 fragment helpers ===================
// A (16x32 bf16, row-major, stride lda): lane<16 -> M=lane, K = k0+{0..7, 16..23}
//                                        lane>=16 -> M=lane-16, K = k0+{8..15, 24..31}
__device__ __forceinline__ v16bf load_a_bf16(const __bf16* __restrict__ A, int lda,
                                             int m0, int k0, int lane) {
    int m  = m0 + (lane & 15);
    int kk = k0 + ((lane >> 4) << 3);          // +0 or +8
    const v8bf* p = (const v8bf*)(A + (size_t)m * (size_t)lda + (size_t)kk);
    v8bf c0 = p[0];                            // K = kk .. kk+7
    v8bf c1 = p[2];                            // K = kk+16 .. kk+23
    return __builtin_shufflevector(c0, c1, 0,1,2,3,4,5,6,7,8,9,10,11,12,13,14,15);
}
// B (32x16 bf16) fragment from an LDS tile stored as [64][kBStride] (N-major):
// lane<16 -> N=nloc+lane, K={0..7,16..23} ; lane>=16 -> N=nloc+lane-16, K={8..15,24..31}
__device__ __forceinline__ v16bf load_b_lds(const __bf16* Bl, int nloc, int lane) {
    int n  = nloc + (lane & 15);
    int kk = (lane >> 4) << 3;                 // +0 or +8
    const v8bf* p = (const v8bf*)(Bl + n * kBStride + kk);
    v8bf c0 = p[0];
    v8bf c1 = p[2];
    return __builtin_shufflevector(c0, c1, 0,1,2,3,4,5,6,7,8,9,10,11,12,13,14,15);
}
// C/D 16x16 f32: VGPR r -> M = r + (lane<16 ? 0 : 8), N = lane&15
__device__ __forceinline__ void store_c_frag(float* __restrict__ C, int ldc,
                                             int m0, int n0, int lane, v8f acc) {
    int c  = lane & 15;
    int rb = (lane >> 4) << 3;
#pragma unroll
    for (int r = 0; r < 8; ++r)
        C[(size_t)(m0 + rb + r) * (size_t)ldc + (size_t)(n0 + c)] = acc[r];
}
__device__ __forceinline__ void store_c_frag_rowscaled(float* __restrict__ C, int ldc,
                                                       int m0, int n0, int lane, v8f acc,
                                                       const float* __restrict__ rs, int rowBase) {
    int c  = lane & 15;
    int rb = (lane >> 4) << 3;
#pragma unroll
    for (int r = 0; r < 8; ++r) {
        int row = m0 + rb + r;
        C[(size_t)row * (size_t)ldc + (size_t)(n0 + c)] = acc[r] * rs[rowBase + row];
    }
}

// =================== split-bf16 (3-product) GEMM, async-LDS B staging ===================
// C[M,N] = (Ahi+Alo)[M,K] @ (Bhi+Blo)[K,N], dropping lo*lo. B supplied transposed (N-major).
// 256 threads = 8 waves; block tile 256(M) x 64(N); wave tile 32(M) x 64(N); K step 32.
// B tile (hi+lo, 8KB) staged once per block into double-buffered LDS with async copies
// overlapping the 24 v_wmma_f32_16x16x32_bf16 of the current chunk.
__global__ void __launch_bounds__(256)
gemm_bf16x3_wmma(const __bf16* __restrict__ Ahi, const __bf16* __restrict__ Alo,
                 const __bf16* __restrict__ Bthi, const __bf16* __restrict__ Btlo,
                 float* __restrict__ C, int M, int N, int K) {
    (void)M;
    __shared__ __bf16 lds[2][2][64 * kBStride];     // [buf][hi/lo][n*kBStride+k], 20 KB
    int tid  = threadIdx.x;
    int lane = tid & 31;
    int wave = tid >> 5;
    int m0 = blockIdx.y * 256 + wave * 32;
    int n0 = blockIdx.x * 64;

    // Each thread stages one 16B chunk of Bhi and one of Blo per k-chunk.
    int sn = tid >> 2;                 // 0..63 : N row within tile
    int sk = (tid & 3) << 3;           // 0,8,16,24 : K offset (elements)
    const __bf16* gh = Bthi + (size_t)(n0 + sn) * (size_t)K + sk;
    const __bf16* gl = Btlo + (size_t)(n0 + sn) * (size_t)K + sk;
    int ldst = sn * kBStride + sk;

    v8f acc[8];
#pragma unroll
    for (int t = 0; t < 8; ++t) acc[t] = (v8f){};

    // prologue: stage chunk 0
    stage16(gh, &lds[0][0][ldst]);
    stage16(gl, &lds[0][1][ldst]);
    stage_wait();
    __syncthreads();

    int nch = K >> 5;
    for (int c = 0; c < nch; ++c) {
        int cur = c & 1;
        if (c + 1 < nch) {                       // prefetch next chunk into other buffer
            int k = (c + 1) << 5;
            stage16(gh + k, &lds[cur ^ 1][0][ldst]);
            stage16(gl + k, &lds[cur ^ 1][1][ldst]);
        }
        int k0 = c << 5;
        v16bf ah0 = load_a_bf16(Ahi, K, m0,      k0, lane);
        v16bf al0 = load_a_bf16(Alo, K, m0,      k0, lane);
        v16bf ah1 = load_a_bf16(Ahi, K, m0 + 16, k0, lane);
        v16bf al1 = load_a_bf16(Alo, K, m0 + 16, k0, lane);
#pragma unroll
        for (int t = 0; t < 4; ++t) {
            v16bf bh = load_b_lds(&lds[cur][0][0], t * 16, lane);
            v16bf bl = load_b_lds(&lds[cur][1][0], t * 16, lane);
            acc[t] = __builtin_amdgcn_wmma_f32_16x16x32_bf16(false, al0, false, bh, (short)0, acc[t], false, false);
            acc[t] = __builtin_amdgcn_wmma_f32_16x16x32_bf16(false, ah0, false, bl, (short)0, acc[t], false, false);
            acc[t] = __builtin_amdgcn_wmma_f32_16x16x32_bf16(false, ah0, false, bh, (short)0, acc[t], false, false);
            acc[4 + t] = __builtin_amdgcn_wmma_f32_16x16x32_bf16(false, al1, false, bh, (short)0, acc[4 + t], false, false);
            acc[4 + t] = __builtin_amdgcn_wmma_f32_16x16x32_bf16(false, ah1, false, bl, (short)0, acc[4 + t], false, false);
            acc[4 + t] = __builtin_amdgcn_wmma_f32_16x16x32_bf16(false, ah1, false, bh, (short)0, acc[4 + t], false, false);
        }
        stage_wait();                             // next buffer landed
        __syncthreads();                          // everyone done reading current buffer
    }
#pragma unroll
    for (int t = 0; t < 4; ++t) {
        store_c_frag(C, N, m0,      n0 + 16 * t, lane, acc[t]);
        store_c_frag(C, N, m0 + 16, n0 + 16 * t, lane, acc[4 + t]);
    }
}

// =================== exact-f32 per-graph aggregation (16x16x4) ===================
__device__ __forceinline__ v2f load_a_f32(const float* __restrict__ A, int lda,
                                          int m0, int k0, int lane) {
    int r  = lane & 15;
    int kk = (lane >> 4) << 1;
    const float* p = A + (size_t)(m0 + r) * (size_t)lda + (size_t)(k0 + kk);
    v2f a; a.x = p[0]; a.y = p[1];
    return a;
}
__device__ __forceinline__ v2f load_b_f32(const float* __restrict__ B, int ldb,
                                          int k0, int n0, int lane) {
    int c  = lane & 15;
    int kk = (lane >> 4) << 1;
    const float* p = B + (size_t)(k0 + kk) * (size_t)ldb + (size_t)(n0 + c);
    v2f b; b.x = p[0]; b.y = p[(size_t)ldb];
    return b;
}

// Out[g*64+m, n] = rs_in[g*64+m] * sum_s Aadj[g][m][s] * H[g*64+s, n]
__global__ void __launch_bounds__(128)
graph_agg_wmma(const float* __restrict__ H, const float* __restrict__ Aadj,
               const float* __restrict__ rs_in, float* __restrict__ Out, int F) {
    int lane = threadIdx.x & 31;
    int wave = threadIdx.x >> 5;
    int g  = blockIdx.y;
    int n0 = blockIdx.x * 64;
    int m0 = wave * 16;
    const float* Ag = Aadj + ((size_t)g << 12);
    const float* Hg = H + (size_t)g * kNPG * (size_t)F;
    float* Og = Out + (size_t)g * kNPG * (size_t)F;
    v8f acc0 = {}, acc1 = {}, acc2 = {}, acc3 = {};
#pragma unroll 4
    for (int k = 0; k < kNPG; k += 4) {
        v2f a  = load_a_f32(Ag, kNPG, m0, k, lane);
        v2f b0 = load_b_f32(Hg, F, k, n0,      lane);
        v2f b1 = load_b_f32(Hg, F, k, n0 + 16, lane);
        v2f b2 = load_b_f32(Hg, F, k, n0 + 32, lane);
        v2f b3 = load_b_f32(Hg, F, k, n0 + 48, lane);
        acc0 = __builtin_amdgcn_wmma_f32_16x16x4_f32(false, a, false, b0, (short)0, acc0, false, false);
        acc1 = __builtin_amdgcn_wmma_f32_16x16x4_f32(false, a, false, b1, (short)0, acc1, false, false);
        acc2 = __builtin_amdgcn_wmma_f32_16x16x4_f32(false, a, false, b2, (short)0, acc2, false, false);
        acc3 = __builtin_amdgcn_wmma_f32_16x16x4_f32(false, a, false, b3, (short)0, acc3, false, false);
    }
    int nodeBase = g * kNPG;
    store_c_frag_rowscaled(Og, F, m0, n0,      lane, acc0, rs_in, nodeBase);
    store_c_frag_rowscaled(Og, F, m0, n0 + 16, lane, acc1, rs_in, nodeBase);
    store_c_frag_rowscaled(Og, F, m0, n0 + 32, lane, acc2, rs_in, nodeBase);
    store_c_frag_rowscaled(Og, F, m0, n0 + 48, lane, acc3, rs_in, nodeBase);
}

// =================== degree + adjacency build ===================
__global__ void fill_zero_kernel(float* __restrict__ p, int n) {
    int i = blockIdx.x * blockDim.x + threadIdx.x;
    if (i < n) p[i] = 0.f;
}

__global__ void build_deg_adj_kernel(const int* __restrict__ src, const int* __restrict__ dst,
                                     const float* __restrict__ ew,
                                     float* __restrict__ degout, float* __restrict__ degin,
                                     float* __restrict__ Aadj) {
    int e = blockIdx.x * blockDim.x + threadIdx.x;
    if (e >= kEdges) return;
    int s = src[e], d = dst[e];
    atomicAdd(&degout[s], 1.0f);
    atomicAdd(&degin[d], 1.0f);
    int g = d >> 6;  // edges are within-graph
    atomicAdd(&Aadj[((size_t)g << 12) + (size_t)((d & 63) << 6) + (size_t)(s & 63)], ew[e]);
}

__global__ void finalize_rs_kernel(float* __restrict__ degout, float* __restrict__ degin) {
    int i = blockIdx.x * blockDim.x + threadIdx.x;
    if (i >= kNodes) return;
    float o = degout[i]; degout[i] = rsqrtf(o < 1.f ? 1.f : o);
    float d = degin[i];  degin[i]  = rsqrtf(d < 1.f ? 1.f : d);
}

// =================== bf16 split / transpose prep ===================
__global__ void split_rowscale_kernel(const float* __restrict__ X, const float* __restrict__ rs,
                                      __bf16* __restrict__ Hi, __bf16* __restrict__ Lo,
                                      int rowLen, long long total) {
    long long i = (long long)blockIdx.x * blockDim.x + threadIdx.x;
    if (i >= total) return;
    float x = X[i];
    if (rs) x *= rs[i / rowLen];
    __bf16 h, l; split_bf16(x, h, l);
    Hi[i] = h; Lo[i] = l;
}

// W[K,N] row-major f32 -> Thi/Tlo[N,K] bf16 (transposed, N-major)
__global__ void split_transpose_kernel(const float* __restrict__ W,
                                       __bf16* __restrict__ Thi, __bf16* __restrict__ Tlo,
                                       int K, int N) {
    long long i = (long long)blockIdx.x * blockDim.x + threadIdx.x;
    if (i >= (long long)K * N) return;
    int k = (int)(i / N);
    int n = (int)(i % N);
    float x = W[i];
    __bf16 h, l; split_bf16(x, h, l);
    Thi[(size_t)n * K + k] = h;
    Tlo[(size_t)n * K + k] = l;
}

// =================== readout / norm kernels ===================
__global__ void mean_readout_kernel(const float* __restrict__ X, float* __restrict__ cat,
                                    int F, int catOff) {
    int c = blockIdx.x * blockDim.x + threadIdx.x;
    int g = blockIdx.y;
    if (c >= F) return;
    const float* p = X + (size_t)g * kNPG * (size_t)F + c;
    float s = 0.f;
#pragma unroll 4
    for (int r = 0; r < kNPG; ++r) s += p[(size_t)r * F];
    cat[(size_t)g * kCat + catOff + c] = s * (1.0f / kNPG);
}

// GraphNorm + LeakyReLU; writes bf16 hi/lo (if given) + graph-mean readout into cat
__global__ void graphnorm_leaky_readout_kernel(const float* __restrict__ X,
        const float* __restrict__ alpha, const float* __restrict__ gamma,
        const float* __restrict__ beta,
        __bf16* __restrict__ Hi, __bf16* __restrict__ Lo,
        float* __restrict__ cat, int F, int catOff) {
    int c = blockIdx.x * blockDim.x + threadIdx.x;
    int g = blockIdx.y;
    if (c >= F) return;
    const float* p = X + (size_t)g * kNPG * (size_t)F + c;
    float s = 0.f;
#pragma unroll 4
    for (int r = 0; r < kNPG; ++r) s += p[(size_t)r * F];
    float am = alpha[c] * (s * (1.0f / kNPG));
    float v = 0.f;
#pragma unroll 4
    for (int r = 0; r < kNPG; ++r) { float d = p[(size_t)r * F] - am; v += d * d; }
    float inv = rsqrtf(v * (1.0f / kNPG) + kEps);
    float ga = gamma[c], be = beta[c];
    float rsum = 0.f;
#pragma unroll 4
    for (int r = 0; r < kNPG; ++r) {
        float y = leaky_f(ga * (p[(size_t)r * F] - am) * inv + be);
        rsum += y;
        if (Hi) {
            __bf16 h, l; split_bf16(y, h, l);
            size_t off = ((size_t)g * kNPG + r) * (size_t)F + c;
            Hi[off] = h; Lo[off] = l;
        }
    }
    cat[(size_t)g * kCat + catOff + c] = rsum * (1.0f / kNPG);
}

// InstanceNorm1d over each row of [kGraphs, kEmb] + LeakyReLU -> d_out
__global__ void __launch_bounds__(256)
instnorm_leaky_kernel(const float* __restrict__ E, float* __restrict__ out) {
    __shared__ float red[256];
    int t = threadIdx.x;
    int row = blockIdx.x;
    const float* p = E + (size_t)row * kEmb;
    float x0 = p[t], x1 = p[t + 256];
    red[t] = x0 + x1;
    __syncthreads();
    for (int s = 128; s > 0; s >>= 1) { if (t < s) red[t] += red[t + s]; __syncthreads(); }
    float mu = red[0] * (1.0f / kEmb);
    __syncthreads();
    float d0 = x0 - mu, d1 = x1 - mu;
    red[t] = d0 * d0 + d1 * d1;
    __syncthreads();
    for (int s = 128; s > 0; s >>= 1) { if (t < s) red[t] += red[t + s]; __syncthreads(); }
    float inv = rsqrtf(red[0] * (1.0f / kEmb) + kEps);
    out[(size_t)row * kEmb + t]       = leaky_f(d0 * inv);
    out[(size_t)row * kEmb + t + 256] = leaky_f(d1 * inv);
}

// =================== launch ===================
extern "C" void kernel_launch(void* const* d_in, const int* in_sizes, int n_in,
                              void* d_out, int out_size, void* d_ws, size_t ws_size,
                              hipStream_t stream) {
    (void)in_sizes; (void)n_in; (void)out_size; (void)ws_size;
    const float* node_feats = (const float*)d_in[0];
    const float* edge_w     = (const float*)d_in[1];
    const float* W1         = (const float*)d_in[2];
    const float* W2         = (const float*)d_in[3];
    const float* W_emb      = (const float*)d_in[4];
    const float* gn1_a      = (const float*)d_in[5];
    const float* gn1_g      = (const float*)d_in[6];
    const float* gn1_b      = (const float*)d_in[7];
    const float* gn2_a      = (const float*)d_in[8];
    const float* gn2_g      = (const float*)d_in[9];
    const float* gn2_b      = (const float*)d_in[10];
    const int*   esrc       = (const int*)d_in[11];
    const int*   edst       = (const int*)d_in[12];
    // d_in[13] node_gid implied by node >> 6

    // ---- workspace layout (byte-based; every sub-buffer 16B aligned) ----
    char* base = (char*)d_ws;
    auto alloc_f32 = [&](size_t n) { float*  p = (float*) base; base += n * sizeof(float);  return p; };
    auto alloc_bf  = [&](size_t n) { __bf16* p = (__bf16*)base; base += n * sizeof(__bf16); return p; };

    float*  rs_out = alloc_f32(kNodes);
    float*  rs_in  = alloc_f32(kNodes);
    float*  Aadj   = alloc_f32((size_t)kGraphs * kNPG * kNPG);
    __bf16* Xhi    = alloc_bf((size_t)kNodes * kInF);
    __bf16* Xlo    = alloc_bf((size_t)kNodes * kInF);
    __bf16* W1thi  = alloc_bf((size_t)kInF * kHid);
    __bf16* W1tlo  = alloc_bf((size_t)kInF * kHid);
    float*  H1pre  = alloc_f32((size_t)kNodes * kHid);     // reused below as h1 hi/lo
    float*  h1agg  = alloc_f32((size_t)kNodes * kHid);
    __bf16* W2thi  = alloc_bf((size_t)kHid * kH4);
    __bf16* W2tlo  = alloc_bf((size_t)kHid * kH4);
    float*  H2pre  = alloc_f32((size_t)kNodes * kH4);
    float*  h2agg  = alloc_f32((size_t)kNodes * kH4);
    float*  cat    = alloc_f32((size_t)kGraphs * kCat);
    __bf16* cathi  = alloc_bf((size_t)kGraphs * kCat);
    __bf16* catlo  = alloc_bf((size_t)kGraphs * kCat);
    __bf16* Wethi  = alloc_bf((size_t)kCat * kEmb);
    __bf16* Wetlo  = alloc_bf((size_t)kCat * kEmb);
    float*  embt   = alloc_f32((size_t)kGraphs * kEmb);
    // h1 bf16 hi/lo overlay H1pre's storage (free once agg1 has consumed it)
    __bf16* h1hi = (__bf16*)H1pre;
    __bf16* h1lo = h1hi + (size_t)kNodes * kHid;

    // ---- degrees + dense per-graph adjacency ----
    int zcount = kNodes * 2 + kGraphs * kNPG * kNPG;   // rs_out, rs_in, Aadj (contiguous)
    fill_zero_kernel<<<(zcount + 255) / 256, 256, 0, stream>>>(rs_out, zcount);
    build_deg_adj_kernel<<<kEdges / 256, 256, 0, stream>>>(esrc, edst, edge_w, rs_out, rs_in, Aadj);
    finalize_rs_kernel<<<kNodes / 256, 256, 0, stream>>>(rs_out, rs_in);

    // ---- r0 readout ----
    mean_readout_kernel<<<dim3(kInF / 256, kGraphs), 256, 0, stream>>>(node_feats, cat, kInF, 0);

    // ---- prep bf16 operands ----
    {
        long long tX = (long long)kNodes * kInF;
        split_rowscale_kernel<<<(unsigned)((tX + 255) / 256), 256, 0, stream>>>(
            node_feats, rs_out, Xhi, Xlo, kInF, tX);
        long long tW1 = (long long)kInF * kHid;
        split_transpose_kernel<<<(unsigned)((tW1 + 255) / 256), 256, 0, stream>>>(
            W1, W1thi, W1tlo, kInF, kHid);
        long long tW2 = (long long)kHid * kH4;
        split_transpose_kernel<<<(unsigned)((tW2 + 255) / 256), 256, 0, stream>>>(
            W2, W2thi, W2tlo, kHid, kH4);
        long long tWe = (long long)kCat * kEmb;
        split_transpose_kernel<<<(unsigned)((tWe + 255) / 256), 256, 0, stream>>>(
            W_emb, Wethi, Wetlo, kCat, kEmb);
    }

    // ---- layer 1: H1pre = (x*rsqrt(deg_out)) @ W1 (split-bf16), agg, norm ----
    gemm_bf16x3_wmma<<<dim3(kHid / 64, kNodes / 256), 256, 0, stream>>>(
        Xhi, Xlo, W1thi, W1tlo, H1pre, kNodes, kHid, kInF);
    graph_agg_wmma<<<dim3(kHid / 64, kGraphs), 128, 0, stream>>>(H1pre, Aadj, rs_in, h1agg, kHid);
    graphnorm_leaky_readout_kernel<<<dim3(kHid / 256, kGraphs), 256, 0, stream>>>(
        h1agg, gn1_a, gn1_g, gn1_b, h1hi, h1lo, cat, kHid, kInF);

    // ---- layer 2 ----
    gemm_bf16x3_wmma<<<dim3(kH4 / 64, kNodes / 256), 256, 0, stream>>>(
        h1hi, h1lo, W2thi, W2tlo, H2pre, kNodes, kH4, kHid);
    graph_agg_wmma<<<dim3(kH4 / 64, kGraphs), 128, 0, stream>>>(H2pre, Aadj, rs_in, h2agg, kH4);
    graphnorm_leaky_readout_kernel<<<dim3(kH4 / 256, kGraphs), 256, 0, stream>>>(
        h2agg, gn2_a, gn2_g, gn2_b, (__bf16*)nullptr, (__bf16*)nullptr, cat, kH4, kInF + kHid);

    // ---- embedding GEMM + per-row instance norm ----
    {
        long long tc = (long long)kGraphs * kCat;
        split_rowscale_kernel<<<(unsigned)((tc + 255) / 256), 256, 0, stream>>>(
            cat, (const float*)nullptr, cathi, catlo, kCat, tc);
    }
    gemm_bf16x3_wmma<<<dim3(kEmb / 64, kGraphs / 256), 256, 0, stream>>>(
        cathi, catlo, Wethi, Wetlo, embt, kGraphs, kEmb, kCat);
    instnorm_leaky_kernel<<<kGraphs, 256, 0, stream>>>(embt, (float*)d_out);
}
